// HippoModel_72688026517783
// MI455X (gfx1250) — matrix operand order repaired and checked by reading, not compiled, not run on hardware
//
#include <hip/hip_runtime.h>
#include <hip/hip_bf16.h>
#include <math.h>

// ---------------- types / WMMA wrappers ----------------
typedef float  v2f   __attribute__((ext_vector_type(2)));
typedef float  v8f   __attribute__((ext_vector_type(8)));
typedef __bf16 v16bf __attribute__((ext_vector_type(16)));

__device__ __forceinline__ v8f wmma_f32(v2f a, v2f b, v8f c) {
  // V_WMMA_F32_16X16X4_F32 : D = A(16x4 f32) x B(4x16 f32) + C(16x16 f32)
  return __builtin_amdgcn_wmma_f32_16x16x4_f32(false, a, false, b, (short)0, c, false, false);
}
__device__ __forceinline__ v8f wmma_bf16(v16bf a, v16bf b, v8f c) {
  // V_WMMA_F32_16X16X32_BF16
  return __builtin_amdgcn_wmma_f32_16x16x32_bf16(false, a, false, b, (short)0, c, false, false);
}

__device__ __forceinline__ float gelu_exact(float x) {
  return 0.5f * x * (1.0f + erff(x * 0.70710678118654752f));
}

// model dims
#define BB   8
#define TT   4096
#define IND  384
#define MIDD 1024
#define HIDD 128
#define FFND 512
#define OUTD 1024
#define LC   64      // chunk length for the scan
#define NCH  64      // number of chunks (TT/LC)

// ---------------- workspace layout (bytes, all 4KB-multiple sizes) ----------
#define OFF_SEQS  0u                         // 32768*128 f32   = 16777216
#define OFF_POW   16777216u                  // 7 * 128*128 f32 = 458752  (A^T,(A^2)^T..(A^64)^T)
#define OFF_CWS   17235968u                  // 64 * 8*128 f32  = 262144
#define OFF_PWS   17498112u                  // 8*128 f32       = 4096
#define OFF_HFIN  17502208u                  // 8*128 f32       = 4096
#define OFF_W1S   17506304u                  // 384*1024 bf16   = 786432
#define OFF_W2S   18292736u                  // 1024*128 bf16   = 262144
#define OFF_PROJ  18554880u                  // 8*1024 f32      = 32768
#define OFF_NORMH 18587648u                  // 8*1024 f32      = 32768
#define OFF_A1    18620416u                  // 8*512  f32      = 16384
#define OFF_FFNO  18636800u                  // 8*1024 f32      = 32768
#define OFF_NORM2 18669568u                  // 8*1024 f32      = 32768
// total ~18.7 MB

// ============================================================================
// K_prep_w : fp32 weights -> bf16, pre-swizzled into the WMMA B-operand layout
//   B layout as used here (contiguous k, same as A): lanes 0-15 hold K=K0+e, lanes 16-31 K=K0+16+e
// ============================================================================
__global__ void K_prep_w(const float* __restrict__ w1, const float* __restrict__ w2,
                         __bf16* __restrict__ W1s, __bf16* __restrict__ W2s) {
  int idx = blockIdx.x * 256 + threadIdx.x;
  if (idx < IND * MIDD) {                       // enc_w1 [384][1024]
    int k = idx >> 10, n = idx & 1023;
    int ks = k >> 5, kk = k & 31;
    int L = (n & 15) + 16 * (kk >> 4);
    int e = kk & 15;
    int nt = n >> 4;
    W1s[(((ks * 64) + nt) * 32 + L) * 16 + e] = (__bf16)w1[idx];
  } else {
    int j = idx - IND * MIDD;
    if (j < MIDD * HIDD) {                      // enc_w2 [1024][128]
      int k = j >> 7, n = j & 127;
      int ks = k >> 5, kk = k & 31;
      int L = (n & 15) + 16 * (kk >> 4);
      int e = kk & 15;
      int nt = n >> 4;
      W2s[((ks * 8 + nt) * 32 + L) * 16 + e] = (__bf16)w2[j];
    }
  }
}

// ============================================================================
// K_pow : single WG. pow[i] = (A^(2^i))^T  (row-major, 128x128), i = 0..6
//   (M^2)^T = (M^T)^2, so we square the transposed matrices directly with
//   fp32 WMMA 16x16x4.
// ============================================================================
__global__ void K_pow(const float* __restrict__ A, float* __restrict__ pw) {
  int tid = threadIdx.x, lane = tid & 31, w = tid >> 5;
  // stage 0: pw[0] = A^T
  for (int idx = tid; idx < 128 * 128; idx += 256) {
    int m = idx >> 7, k = idx & 127;
    pw[k * 128 + m] = A[idx];
  }
  __threadfence(); __syncthreads();

  int col = w * 16 + (lane & 15);
  for (int i = 0; i < 6; ++i) {
    const float* src = pw + i * 16384;
    float*       dst = pw + (i + 1) * 16384;
    for (int mt = 0; mt < 8; ++mt) {
      v8f acc[4]; acc[0] = {}; acc[1] = {}; acc[2] = {}; acc[3] = {};
#pragma unroll
      for (int ks = 0; ks < 32; ++ks) {
        v2f a = *(const v2f*)(src + (mt * 16 + (lane & 15)) * 128 + 4 * ks + 2 * (lane >> 4));
        v2f b;
        b.x = src[(4 * ks     + 2 * (lane >> 4)) * 128 + col];
        b.y = src[(4 * ks + 1 + 2 * (lane >> 4)) * 128 + col];
        acc[ks & 3] = wmma_f32(a, b, acc[ks & 3]);
      }
      v8f c = acc[0] + acc[1] + acc[2] + acc[3];
#pragma unroll
      for (int v = 0; v < 8; ++v)
        dst[(mt * 16 + v + 8 * (lane >> 4)) * 128 + col] = c[v];
    }
    __threadfence(); __syncthreads();
  }
}

// ============================================================================
// K_enc : fused encoder. Each WG handles a 32-row stripe of the 32768 tokens.
//   GEMM1 (K=384, bf16 WMMA) + bias + exact GELU  ->  LDS chunk (A layout)
//   GEMM2 (K=1024 accumulated over 4 phases)      ->  seqs_bx fp32
// ============================================================================
__global__ void K_enc(const float* __restrict__ x, const float* __restrict__ b1,
                      const float* __restrict__ b2, const __bf16* __restrict__ W1s,
                      const __bf16* __restrict__ W2s, float* __restrict__ seqs) {
  __shared__ __bf16 Xs[2 * 12 * 32 * 16];  // 24KB: x stripe, A layout (2 mtiles x 12 ksteps)
  __shared__ __bf16 Hc[2 * 8 * 32 * 16];   // 16KB: GELU(h_mid) phase chunk, A layout

  const int tid = threadIdx.x, lane = tid & 31, w = tid >> 5;
  const int row0 = blockIdx.x * 32;

  // stage x stripe -> LDS (bf16, A-operand layout)
  for (int idx = tid; idx < 32 * IND; idx += 256) {
    int r = idx / IND, k = idx - r * IND;
    float val = x[(row0 + r) * IND + k];
    int mt = r >> 4, m = r & 15;
    int ks = k >> 5, kk = k & 31;
    int g = kk >> 4;
    int e = kk & 15;
    int L = m + 16 * g;
    Xs[((mt * 12 + ks) * 32 + L) * 16 + e] = (__bf16)val;
  }
  __syncthreads();

  const v16bf* w1v = (const v16bf*)W1s;
  const v16bf* w2v = (const v16bf*)W2s;

  v8f acc2[2];
  {
    float bv2 = b2[w * 16 + (lane & 15)];
    v8f t{};
#pragma unroll
    for (int e = 0; e < 8; ++e) t[e] = bv2;
    acc2[0] = t; acc2[1] = t;
  }

  for (int p = 0; p < 4; ++p) {                       // 4 N-phases of 256 cols
    v8f acc1[2][2];
#pragma unroll
    for (int ntl = 0; ntl < 2; ++ntl) {
      float bv = b1[p * 256 + w * 32 + ntl * 16 + (lane & 15)];
      v8f t{};
#pragma unroll
      for (int e = 0; e < 8; ++e) t[e] = bv;
      acc1[0][ntl] = t; acc1[1][ntl] = t;
    }
#pragma unroll
    for (int ks = 0; ks < 12; ++ks) {
      v16bf a0 = *(const v16bf*)&Xs[((0 * 12 + ks) * 32 + lane) * 16];
      v16bf a1 = *(const v16bf*)&Xs[((1 * 12 + ks) * 32 + lane) * 16];
#pragma unroll
      for (int ntl = 0; ntl < 2; ++ntl) {
        int nt = p * 16 + w * 2 + ntl;
        v16bf bmat = w1v[(ks * 64 + nt) * 32 + lane];
        acc1[0][ntl] = wmma_bf16(a0, bmat, acc1[0][ntl]);
        acc1[1][ntl] = wmma_bf16(a1, bmat, acc1[1][ntl]);
      }
    }
    // GELU -> LDS chunk (A layout for GEMM2)
#pragma unroll
    for (int mt = 0; mt < 2; ++mt)
#pragma unroll
      for (int ntl = 0; ntl < 2; ++ntl)
#pragma unroll
        for (int v = 0; v < 8; ++v) {
          float gv = gelu_exact(acc1[mt][ntl][v]);
          int c_local = (w * 2 + ntl) * 16 + (lane & 15);
          int m_local = mt * 16 + v + 8 * (lane >> 4);
          int ks2 = c_local >> 5, kk = c_local & 31;
          int g2 = kk >> 4;
          int e = kk & 15;
          int L2 = (m_local & 15) + 16 * g2;
          Hc[((mt * 8 + ks2) * 32 + L2) * 16 + e] = (__bf16)gv;
        }
    __syncthreads();
    // partial GEMM2 over this 256-wide K chunk
#pragma unroll
    for (int ks2 = 0; ks2 < 8; ++ks2) {
      int ksg = p * 8 + ks2;
      v16bf bmat = w2v[(ksg * 8 + w) * 32 + lane];
      v16bf a0 = *(const v16bf*)&Hc[((0 * 8 + ks2) * 32 + lane) * 16];
      v16bf a1 = *(const v16bf*)&Hc[((1 * 8 + ks2) * 32 + lane) * 16];
      acc2[0] = wmma_bf16(a0, bmat, acc2[0]);
      acc2[1] = wmma_bf16(a1, bmat, acc2[1]);
    }
    __syncthreads();
  }
  // store seqs_bx (fp32)
#pragma unroll
  for (int mt = 0; mt < 2; ++mt)
#pragma unroll
    for (int v = 0; v < 8; ++v) {
      int rowg = row0 + mt * 16 + v + 8 * (lane >> 4);
      seqs[rowg * 128 + w * 16 + (lane & 15)] = acc2[mt][v];
    }
}

// ============================================================================
// K_chunk : 64 WGs; chunk j runs the local 64-step scan c = A*c + x_t (c0=0)
//           with fp32 WMMA; captures per-batch prefix p_b at t == seq_len-1.
// ============================================================================
__global__ void K_chunk(const float* __restrict__ seqs, const float* __restrict__ AT,
                        const int* __restrict__ sl, float* __restrict__ Cws,
                        float* __restrict__ Pws) {
  __shared__ float hbuf[2][16 * 128];
  const int tid = threadIdx.x, lane = tid & 31, w = tid >> 5;
  const int j = blockIdx.x;
  const int col = w * 16 + (lane & 15);

  for (int i = tid; i < 2048; i += 256) hbuf[0][i] = 0.0f;

  // B operand = A^T slice for this wave's 16 columns (held in registers)
  float Bv[64];
#pragma unroll
  for (int ks = 0; ks < 32; ++ks) {
    Bv[2 * ks]     = AT[(4 * ks     + 2 * (lane >> 4)) * 128 + col];
    Bv[2 * ks + 1] = AT[(4 * ks + 1 + 2 * (lane >> 4)) * 128 + col];
  }
  int s[8];
#pragma unroll
  for (int b = 0; b < 8; ++b) s[b] = sl[b];

  __syncthreads();
  int p = 0;
#pragma unroll 1
  for (int tl = 0; tl < LC; ++tl) {
    const int t = j * LC + tl;
    // C init = x_t tile (rows 8..15 and lanes>=16 are zero padding)
    v8f acc0{}; v8f acc1{}; v8f acc2{}; v8f acc3{};
#pragma unroll
    for (int v = 0; v < 8; ++v) {
      float xv = 0.0f;
      if (lane < 16) xv = seqs[(v * TT + t) * 128 + col];
      acc0[v] = xv;
    }
    if (tl < LC - 1 && lane < 16)
      __builtin_prefetch(&seqs[(0 * TT + t + 1) * 128 + col], 0, 0);
#pragma unroll
    for (int ks = 0; ks < 32; ++ks) {
      v2f a = *(const v2f*)&hbuf[p][(lane & 15) * 128 + 4 * ks + 2 * (lane >> 4)];
      v2f b; b.x = Bv[2 * ks]; b.y = Bv[2 * ks + 1];
      switch (ks & 3) {
        case 0: acc0 = wmma_f32(a, b, acc0); break;
        case 1: acc1 = wmma_f32(a, b, acc1); break;
        case 2: acc2 = wmma_f32(a, b, acc2); break;
        default: acc3 = wmma_f32(a, b, acc3); break;
      }
    }
    v8f hn = acc0 + acc1 + acc2 + acc3;
    // capture local prefix at the cut position
#pragma unroll
    for (int b = 0; b < 8; ++b)
      if (t == s[b] - 1 && lane < 16) Pws[b * 128 + col] = hn[b];
    // write new state
#pragma unroll
    for (int v = 0; v < 8; ++v)
      hbuf[1 - p][(v + 8 * (lane >> 4)) * 128 + col] = hn[v];
    __syncthreads();
    p ^= 1;
  }
  // store c_j (rows 0..7)
  for (int i = tid; i < 1024; i += 256)
    Cws[j * 1024 + i] = hbuf[p][(i >> 7) * 128 + (i & 127)];
}

// ============================================================================
// K_comb : single WG. H_{j+1} = A^64 H_j + c_j  (fp32 WMMA, 64 steps).
//   Before combining chunk j, finalize every batch whose cut lies in chunk j:
//   h_b = A^(r+1) H_j[b] + p_b, applying binary powers (<=7 coalesced matvecs).
// ============================================================================
__global__ void K_comb(const float* __restrict__ pw, const float* __restrict__ Cws,
                       const float* __restrict__ Pws, const int* __restrict__ sl,
                       float* __restrict__ hfin) {
  __shared__ float hbuf[2][16 * 128];
  __shared__ float vbuf[2][128];
  const int tid = threadIdx.x, lane = tid & 31, w = tid >> 5;
  const int col = w * 16 + (lane & 15);
  const float* A64T = pw + 6 * 16384;

  for (int i = tid; i < 2048; i += 256) hbuf[0][i] = 0.0f;

  float Bv[64];
#pragma unroll
  for (int ks = 0; ks < 32; ++ks) {
    Bv[2 * ks]     = A64T[(4 * ks     + 2 * (lane >> 4)) * 128 + col];
    Bv[2 * ks + 1] = A64T[(4 * ks + 1 + 2 * (lane >> 4)) * 128 + col];
  }
  int s[8];
#pragma unroll
  for (int b = 0; b < 8; ++b) s[b] = sl[b];

  __syncthreads();
  int p = 0;
#pragma unroll 1
  for (int j = 0; j < NCH; ++j) {
    // ---- finalize batches whose last update lands in chunk j ----
#pragma unroll 1
    for (int b = 0; b < 8; ++b) {
      int jb = (s[b] - 1) >> 6;
      if (jb == j) {
        if (tid < 128) vbuf[0][tid] = hbuf[p][b * 128 + tid];
        __syncthreads();
        int rp1 = s[b] - (jb << 6);   // 1..64
        int cur = 0;
#pragma unroll 1
        for (int i = 0; i < 7; ++i) {
          if ((rp1 >> i) & 1) {
            const float* Mi = pw + i * 16384;
            float accv = 0.0f;
            if (tid < 128) {
#pragma unroll 4
              for (int k = 0; k < 128; ++k) accv = fmaf(Mi[k * 128 + tid], vbuf[cur][k], accv);
            }
            __syncthreads();
            if (tid < 128) vbuf[1 - cur][tid] = accv;
            __syncthreads();
            cur ^= 1;
          }
        }
        if (tid < 128) hfin[b * 128 + tid] = vbuf[cur][tid] + Pws[b * 128 + tid];
        __syncthreads();
      }
    }
    // ---- combine: H = H @ (A^64)^T + c_j ----
    v8f acc0{}; v8f acc1{}; v8f acc2{}; v8f acc3{};
#pragma unroll
    for (int v = 0; v < 8; ++v) {
      float xv = 0.0f;
      if (lane < 16) xv = Cws[j * 1024 + v * 128 + col];
      acc0[v] = xv;
    }
#pragma unroll
    for (int ks = 0; ks < 32; ++ks) {
      v2f a = *(const v2f*)&hbuf[p][(lane & 15) * 128 + 4 * ks + 2 * (lane >> 4)];
      v2f b; b.x = Bv[2 * ks]; b.y = Bv[2 * ks + 1];
      switch (ks & 3) {
        case 0: acc0 = wmma_f32(a, b, acc0); break;
        case 1: acc1 = wmma_f32(a, b, acc1); break;
        case 2: acc2 = wmma_f32(a, b, acc2); break;
        default: acc3 = wmma_f32(a, b, acc3); break;
      }
    }
    v8f hn = acc0 + acc1 + acc2 + acc3;
#pragma unroll
    for (int v = 0; v < 8; ++v)
      hbuf[1 - p][(v + 8 * (lane >> 4)) * 128 + col] = hn[v];
    __syncthreads();
    p ^= 1;
  }
}

// ============================================================================
// decoder helpers (8 rows only — bandwidth/latency trivial)
// ============================================================================
__global__ void k_dot(const float* __restrict__ in, const float* __restrict__ wgt,
                      const float* __restrict__ bias, const float* __restrict__ resid,
                      float* __restrict__ out, int K, int N, int act) {
  int gid = blockIdx.x * 256 + threadIdx.x;
  if (gid >= 8 * N) return;
  int r = gid / N, n = gid - r * N;
  float acc = bias[n];
#pragma unroll 4
  for (int k = 0; k < K; ++k) acc = fmaf(in[r * K + k], wgt[k * N + n], acc);
  if (act) acc = gelu_exact(acc);
  if (resid) acc += resid[gid];
  out[gid] = acc;
}

__global__ void k_ln(const float* __restrict__ in, const float* __restrict__ g,
                     const float* __restrict__ b, float* __restrict__ out) {
  __shared__ float s1[256], s2[256];
  int r = blockIdx.x, tid = threadIdx.x;
  float a = 0.0f, q = 0.0f;
  for (int c = tid; c < 1024; c += 256) {
    float x = in[r * 1024 + c];
    a += x; q += x * x;
  }
  s1[tid] = a; s2[tid] = q; __syncthreads();
  for (int off = 128; off > 0; off >>= 1) {
    if (tid < off) { s1[tid] += s1[tid + off]; s2[tid] += s2[tid + off]; }
    __syncthreads();
  }
  float mu = s1[0] * (1.0f / 1024.0f);
  float var = s2[0] * (1.0f / 1024.0f) - mu * mu;
  float rs = rsqrtf(var + 1e-5f);
  for (int c = tid; c < 1024; c += 256) {
    float x = in[r * 1024 + c];
    out[r * 1024 + c] = (x - mu) * rs * g[c] + b[c];
  }
}

// ============================================================================
extern "C" void kernel_launch(void* const* d_in, const int* in_sizes, int n_in,
                              void* d_out, int out_size, void* d_ws, size_t ws_size,
                              hipStream_t stream) {
  (void)in_sizes; (void)n_in; (void)out_size; (void)ws_size;
  const float* x      = (const float*)d_in[0];
  const int*   sl     = (const int*)d_in[1];
  const float* A      = (const float*)d_in[2];
  const float* enc_w1 = (const float*)d_in[3];
  const float* enc_b1 = (const float*)d_in[4];
  const float* enc_w2 = (const float*)d_in[5];
  const float* enc_b2 = (const float*)d_in[6];
  const float* proj_w = (const float*)d_in[7];
  const float* proj_b = (const float*)d_in[8];
  const float* ln1_g  = (const float*)d_in[9];
  const float* ln1_b  = (const float*)d_in[10];
  const float* ffn_w1 = (const float*)d_in[11];
  const float* ffn_b1 = (const float*)d_in[12];
  const float* ffn_w2 = (const float*)d_in[13];
  const float* ffn_b2 = (const float*)d_in[14];
  const float* ln2_g  = (const float*)d_in[15];
  const float* ln2_b  = (const float*)d_in[16];
  const float* out_w  = (const float*)d_in[17];
  const float* out_b  = (const float*)d_in[18];

  char* ws = (char*)d_ws;
  float*  seqs  = (float*)(ws + OFF_SEQS);
  float*  pw    = (float*)(ws + OFF_POW);
  float*  Cws   = (float*)(ws + OFF_CWS);
  float*  Pws   = (float*)(ws + OFF_PWS);
  float*  hfin  = (float*)(ws + OFF_HFIN);
  __bf16* W1s   = (__bf16*)(ws + OFF_W1S);
  __bf16* W2s   = (__bf16*)(ws + OFF_W2S);
  float*  proj  = (float*)(ws + OFF_PROJ);
  float*  normh = (float*)(ws + OFF_NORMH);
  float*  a1    = (float*)(ws + OFF_A1);
  float*  ffno  = (float*)(ws + OFF_FFNO);
  float*  norm2 = (float*)(ws + OFF_NORM2);

  K_prep_w<<<2048, 256, 0, stream>>>(enc_w1, enc_w2, W1s, W2s);
  K_pow<<<1, 256, 0, stream>>>(A, pw);
  K_enc<<<1024, 256, 0, stream>>>(x, enc_b1, enc_b2, W1s, W2s, seqs);
  K_chunk<<<NCH, 256, 0, stream>>>(seqs, pw /* A^T */, sl, Cws, Pws);
  K_comb<<<1, 256, 0, stream>>>(pw, Cws, Pws, sl, hfin);

  k_dot<<<32, 256, 0, stream>>>(hfin, proj_w, proj_b, nullptr, proj, 128, 1024, 0);
  k_ln<<<8, 256, 0, stream>>>(proj, ln1_g, ln1_b, normh);
  k_dot<<<16, 256, 0, stream>>>(normh, ffn_w1, ffn_b1, nullptr, a1, 1024, 512, 1);
  k_dot<<<32, 256, 0, stream>>>(a1, ffn_w2, ffn_b2, proj, ffno, 512, 1024, 0);
  k_ln<<<8, 256, 0, stream>>>(ffno, ln2_g, ln2_b, norm2);
  k_dot<<<32, 256, 0, stream>>>(norm2, out_w, out_b, nullptr, (float*)d_out, 1024, 1024, 0);
}
